// BiFormerLiteBlock_32942399160788
// MI455X (gfx1250) — compile-verified, matrix operand order checked by reading, not run on hardware
//
#include <hip/hip_runtime.h>
#include <hip/hip_bf16.h>

typedef _Float16 half_t;
typedef __attribute__((ext_vector_type(16))) _Float16 v16h;
typedef __attribute__((ext_vector_type(8)))  float    v8f;
typedef __attribute__((ext_vector_type(4)))  unsigned u32x4;
typedef __attribute__((ext_vector_type(8)))  int      i32x8;
typedef __attribute__((ext_vector_type(4)))  int      i32x4;

union F16x16 { v16h v; uint4 q[2]; _Float16 h[16]; };
union H8     { uint4 q; _Float16 h[8]; };

#define T_TOK 32768   // B*H*W
#define C_CH  256
#define HW    4096

#if defined(__has_builtin)
#if __has_builtin(__builtin_amdgcn_tensor_load_to_lds) && __has_builtin(__builtin_amdgcn_s_wait_tensorcnt)
#define HAS_TDM 1
#endif
#endif
#ifndef HAS_TDM
#define HAS_TDM 0
#endif

// ---------------------------------------------------------------------------
// cross-lane reductions via ds_swizzle (group-of-32 xor mode, imm patterns)
// ---------------------------------------------------------------------------
template<int IMM>
__device__ inline float swzf(float v) {
  return __int_as_float(__builtin_amdgcn_ds_swizzle(__float_as_int(v), IMM));
}
__device__ inline float red_max16(float v) {
  v = fmaxf(v, swzf<0x041F>(v));
  v = fmaxf(v, swzf<0x081F>(v));
  v = fmaxf(v, swzf<0x101F>(v));
  v = fmaxf(v, swzf<0x201F>(v));
  return v;
}
__device__ inline float red_sum16(float v) {
  v += swzf<0x041F>(v);
  v += swzf<0x081F>(v);
  v += swzf<0x101F>(v);
  v += swzf<0x201F>(v);
  return v;
}
__device__ inline float red_sum32(float v) {
  v = red_sum16(v);
  v += swzf<0x401F>(v);
  return v;
}

// ---------------------------------------------------------------------------
// TDM: async 2D fp16 tile (rows x 32 halves) global->LDS, 8-half row pad
// (LDS pitch = 40 halves). Issued by one wave; tracked by TENSORcnt.
// 6-arg toolchain form: (g0, g1, g2, g3, g_ext, cpol)
// ---------------------------------------------------------------------------
#if HAS_TDM
__device__ inline unsigned lds_off(const void* p) {
  return (unsigned)(unsigned long long)p;   // low 32 bits of flat addr = LDS byte addr
}
__device__ inline void tdm_load_tile(const half_t* gptr, unsigned lds_addr,
                                     int rows, int stride_halves) {
  unsigned long long ga = (unsigned long long)gptr;
  u32x4 g0;
  g0[0] = 1u;                                   // count=1, user descriptor
  g0[1] = lds_addr;                             // LDS byte address
  g0[2] = (unsigned)ga;                         // global_addr[31:0]
  g0[3] = (unsigned)((ga >> 32) & 0x01FFFFFFu) | (2u << 30);  // addr[56:32] | type=2
  i32x8 g1;
  // data_size=1(2B) | pad_enable | pad_interval=3(16 dwords) | pad_amount=3(4 dwords)
  g1[0] = (int)((1u << 16) | (1u << 20) | (3u << 22) | (3u << 25));
  g1[1] = (int)(32u << 16);                     // tensor_dim0 = 32 (lo16 @ bits63:48)
  g1[2] = (int)((unsigned)rows << 16);          // tensor_dim1 lo16 @ bits95:80
  g1[3] = (int)(32u << 16);                     // tile_dim0 = 32 @ bits127:112
  g1[4] = rows;                                 // tile_dim1 @ bits143:128
  g1[5] = stride_halves;                        // tensor_dim0_stride lo32
  g1[6] = 0;
  g1[7] = 0;
  i32x4 g2 = {0, 0, 0, 0};
  i32x4 g3 = {0, 0, 0, 0};
  i32x8 gx = {0, 0, 0, 0, 0, 0, 0, 0};
  __builtin_amdgcn_tensor_load_to_lds(g0, g1, g2, g3, gx, 0);
}
#endif

// ---------------------------------------------------------------------------
// LayerNorm over channels of (B,C,H,W) fp32 -> token-major [T,256] fp16
// ---------------------------------------------------------------------------
__global__ __launch_bounds__(256)
void ln_kernel(const float* __restrict__ X, const float* __restrict__ w,
               const float* __restrict__ b, half_t* __restrict__ Y) {
  int tid = threadIdx.x;
  int pl = tid & 63, cg = tid >> 6;
  long t = (long)blockIdx.x * 64 + pl;
  int bb = (int)(t >> 12); int p = (int)(t & 4095);
  const float* xp = X + ((long)bb * C_CH + cg * 64) * HW + p;
  float s = 0.f, ss = 0.f;
#pragma unroll 16
  for (int c = 0; c < 64; c++) { float v = xp[(long)c * HW]; s += v; ss += v * v; }
  __shared__ float Ssum[4][64], Ssq[4][64];
  Ssum[cg][pl] = s; Ssq[cg][pl] = ss;
  __syncthreads();
  float ts = 0.f, tss = 0.f;
#pragma unroll
  for (int i = 0; i < 4; i++) { ts += Ssum[i][pl]; tss += Ssq[i][pl]; }
  float mean = ts * (1.f / 256.f);
  float var  = tss * (1.f / 256.f) - mean * mean;
  float rstd = rsqrtf(var + 1e-6f);
  half_t* yp = Y + t * C_CH + cg * 64;
  const float* wp = w + cg * 64; const float* bp = b + cg * 64;
#pragma unroll 16
  for (int c = 0; c < 64; c++) {
    float v = xp[(long)c * HW];
    yp[c] = (half_t)(wp[c] * (v - mean) * rstd + bp[c]);
  }
}

// ---------------------------------------------------------------------------
// fp32 -> fp16 weight pack
// ---------------------------------------------------------------------------
__global__ void f2h_kernel(const float* __restrict__ s, half_t* __restrict__ d, int n) {
  int i = blockIdx.x * 256 + threadIdx.x;
  if (i < n) d[i] = (half_t)s[i];
}

// ---------------------------------------------------------------------------
// WMMA GEMM: Y[M,N] = A[M,K] * W[N,K]^T + bias
// 256 thr / 8 waves, block tile 128x64, wave tile 32x32 (2x2 frags), BK=32.
// Double-buffered LDS; TDM async tile loads when no row gather.
// EPI: 0 f16 store | 1 gelu->f16 | 2 fp32 residual add (B,C,H,W) | 3 final out
// ---------------------------------------------------------------------------
#define BM 128
#define BN 64
#define BK 32
#define LDS_A 40
#define LDS_W 40

template<int EPI>
__global__ __launch_bounds__(256)
void gemm_kernel(const half_t* __restrict__ A, int lda,
                 const half_t* __restrict__ W, int ldw,
                 const float* __restrict__ bias,
                 half_t* __restrict__ Y, int ldy,
                 float* __restrict__ R, float* __restrict__ Om,
                 const int* __restrict__ rowidx, int K) {
  __shared__ half_t As[2][BM * LDS_A];
  __shared__ half_t Ws[2][BN * LDS_W];
  int tid = threadIdx.x;
  int wid = tid >> 5, lane = tid & 31;
  int wm = wid >> 1, wn = wid & 1;
  int bm = blockIdx.x * BM, bn = blockIdx.y * BN;
  int g8 = (lane >> 4) * 8, l15 = lane & 15;
  int nsteps = K / BK;

  v8f zero = {0.f, 0.f, 0.f, 0.f, 0.f, 0.f, 0.f, 0.f};
  v8f acc[2][2];
  acc[0][0] = zero; acc[0][1] = zero; acc[1][0] = zero; acc[1][1] = zero;

  // manual-staging addressing (used for gather path / no-TDM fallback)
  int ar = tid >> 1, ac = (tid & 1) * 16;
  long arow = rowidx ? (long)rowidx[bm + ar] : (long)(bm + ar);
  const half_t* aptr = A + arow * lda + ac;
  int wr = tid >> 2, wc = (tid & 3) * 8;
  const half_t* wptr = W + (long)(bn + wr) * ldw + wc;

#if HAS_TDM
  const bool use_tdm = (rowidx == nullptr);
  if (use_tdm && tid < 32) {   // wave 0 owns TDM issue (EXEC ignored by TDM)
    tdm_load_tile(A + (long)bm * lda, lds_off(&As[0][0]), BM, lda);
    tdm_load_tile(W + (long)bn * ldw, lds_off(&Ws[0][0]), BN, ldw);
  }
#else
  const bool use_tdm = false;
#endif

  for (int s = 0; s < nsteps; s++) {
    int cur = s & 1;
    int k0 = s * BK;
    if (use_tdm) {
#if HAS_TDM
      if (tid < 32) {
        if (s + 1 < nsteps) {
          tdm_load_tile(A + (long)bm * lda + k0 + BK, lds_off(&As[cur ^ 1][0]), BM, lda);
          tdm_load_tile(W + (long)bn * ldw + k0 + BK, lds_off(&Ws[cur ^ 1][0]), BN, ldw);
          __builtin_amdgcn_s_wait_tensorcnt(2);  // current buffers landed
        } else {
          __builtin_amdgcn_s_wait_tensorcnt(0);
        }
      }
#endif
    } else {
      *(uint4*)&As[cur][ar * LDS_A + ac]     = *(const uint4*)(aptr + k0);
      *(uint4*)&As[cur][ar * LDS_A + ac + 8] = *(const uint4*)(aptr + k0 + 8);
      *(uint4*)&Ws[cur][wr * LDS_W + wc]     = *(const uint4*)(wptr + k0);
      if (k0 + BK < K) {
        __builtin_prefetch(aptr + k0 + BK, 0, 0);
        __builtin_prefetch(wptr + k0 + BK, 0, 0);
      }
    }
    __syncthreads();

    F16x16 af[2], bf[2];
#pragma unroll
    for (int i = 0; i < 2; i++) {
      int row = wm * 32 + i * 16 + l15;
      af[i].q[0] = *(const uint4*)&As[cur][row * LDS_A + g8];
      af[i].q[1] = *(const uint4*)&As[cur][row * LDS_A + 16 + g8];
    }
#pragma unroll
    for (int j = 0; j < 2; j++) {
      int row = wn * 32 + j * 16 + l15;
      bf[j].q[0] = *(const uint4*)&Ws[cur][row * LDS_W + g8];
      bf[j].q[1] = *(const uint4*)&Ws[cur][row * LDS_W + 16 + g8];
    }
#pragma unroll
    for (int i = 0; i < 2; i++)
#pragma unroll
      for (int j = 0; j < 2; j++)
        acc[i][j] = __builtin_amdgcn_wmma_f32_16x16x32_f16(
            false, af[i].v, false, bf[j].v, (short)0, acc[i][j], false, false);
    __syncthreads();
  }

  // epilogue
#pragma unroll
  for (int i = 0; i < 2; i++)
#pragma unroll
    for (int j = 0; j < 2; j++) {
      int nloc = bn + wn * 32 + j * 16 + l15;
      float bv = bias ? bias[nloc] : 0.f;
#pragma unroll
      for (int r = 0; r < 8; r++) {
        int mloc = bm + wm * 32 + i * 16 + g8 + r;
        float val = acc[i][j][r] + bv;
        if (EPI == 0) {
          Y[(long)mloc * ldy + nloc] = (half_t)val;
        } else if (EPI == 1) {
          float ge = 0.5f * val * (1.f + erff(val * 0.70710678f));
          Y[(long)mloc * ldy + nloc] = (half_t)ge;
        } else {
          int bb = mloc >> 12, p = mloc & 4095;
          long off = (((long)bb * C_CH + nloc) << 12) + p;
          if (EPI == 2) R[off] += val;
          else          Om[off] = R[off] + val;
        }
      }
    }
}

// ---------------------------------------------------------------------------
// per-token sum of squares: smap[t] = sum_c score[t,c]^2 (one wave per token)
// ---------------------------------------------------------------------------
__global__ __launch_bounds__(256)
void sqsum_kernel(const half_t* __restrict__ S, float* __restrict__ smap) {
  int tid = threadIdx.x; int lane = tid & 31; int wv = tid >> 5;
  long t = (long)blockIdx.x * 8 + wv;
  const half_t* p = S + t * C_CH + lane * 8;
  H8 h; h.q = *(const uint4*)p;
  float s = 0.f;
#pragma unroll
  for (int i = 0; i < 8; i++) { float v = (float)h.h[i]; s += v * v; }
  s = red_sum32(s);
  if (lane == 0) smap[t] = s;
}

// ---------------------------------------------------------------------------
// top-64 per batch (iterative argmax); grid = B blocks of 256
// ---------------------------------------------------------------------------
__global__ __launch_bounds__(256)
void topk_kernel(const float* __restrict__ smap, int* __restrict__ gidx) {
  int b = blockIdx.x, tid = threadIdx.x;
  __shared__ float vals[HW];
  __shared__ float rv[256];
  __shared__ int   ri[256];
  for (int i = tid; i < HW; i += 256) vals[i] = smap[(long)b * HW + i];
  __syncthreads();
  for (int it = 0; it < 64; it++) {
    float bv = -1e30f; int bi = 0;
#pragma unroll 4
    for (int i = tid; i < HW; i += 256) { float v = vals[i]; if (v > bv) { bv = v; bi = i; } }
    rv[tid] = bv; ri[tid] = bi;
    __syncthreads();
    for (int s = 128; s > 0; s >>= 1) {
      if (tid < s && rv[tid + s] > rv[tid]) { rv[tid] = rv[tid + s]; ri[tid] = ri[tid + s]; }
      __syncthreads();
    }
    if (tid == 0) { gidx[b * 64 + it] = b * HW + ri[0]; vals[ri[0]] = -1e30f; }
    __syncthreads();
  }
}

// ---------------------------------------------------------------------------
// Attention (local windows & routed-global): Lq=Lk=64, d=64, 4 waves/block.
// ---------------------------------------------------------------------------
__device__ inline int q_token(int mode, int grp, int i) {
  if (mode == 0) {
    int b = grp >> 6, rem = grp & 63, hb = rem >> 3, wb = rem & 7;
    return (b << 12) + ((hb << 3) + (i >> 3)) * 64 + (wb << 3) + (i & 7);
  }
  return ((grp >> 6) << 12) + ((grp & 63) << 6) + i;
}
__device__ inline int kv_row(int mode, int grp, int i) {
  if (mode == 0) return q_token(0, grp, i);
  return ((grp >> 6) << 6) + i;
}

__global__ __launch_bounds__(128)
void attn_kernel(const half_t* __restrict__ Qg, int qld,
                 const half_t* __restrict__ Kg, int kld,
                 const half_t* __restrict__ Vg, int vld,
                 half_t* __restrict__ Og, int mode) {
  __shared__ half_t Qs[64 * 72];
  __shared__ half_t Ks[64 * 72];
  __shared__ half_t Vt[64 * 72];
  __shared__ half_t Ps[64 * 72];

  int tid = threadIdx.x;
  int head = blockIdx.x & 3;
  int grp  = blockIdx.x >> 2;
  int hoff = head * 64;
  int lane = tid & 31, wid = tid >> 5;
  int rb = wid * 16;
  int g8 = (lane >> 4) * 8, l15 = lane & 15;

  // ---- stage Q, K, V^T into LDS ----
  {
    int i = tid >> 1, c0 = (tid & 1) * 32;
    long qt = q_token(mode, grp, i);
    long kr = kv_row(mode, grp, i);
    const half_t* qp = Qg + qt * qld + hoff + c0;
    const half_t* kp = Kg + kr * kld + hoff + c0;
    const half_t* vp = Vg + kr * vld + hoff + c0;
#pragma unroll
    for (int cc = 0; cc < 32; cc += 8) {
      *(uint4*)&Qs[i * 72 + c0 + cc] = *(const uint4*)(qp + cc);
      *(uint4*)&Ks[i * 72 + c0 + cc] = *(const uint4*)(kp + cc);
      H8 h; h.q = *(const uint4*)(vp + cc);
#pragma unroll
      for (int e = 0; e < 8; e++) Vt[(c0 + cc + e) * 72 + i] = h.h[e];
    }
  }
  __syncthreads();

  // ---- S = Q K^T ----
  v8f zero = {0.f, 0.f, 0.f, 0.f, 0.f, 0.f, 0.f, 0.f};
  v8f s[4]; s[0] = zero; s[1] = zero; s[2] = zero; s[3] = zero;
#pragma unroll
  for (int k0 = 0; k0 < 64; k0 += 32) {
    F16x16 af;
    int row = rb + l15;
    af.q[0] = *(const uint4*)&Qs[row * 72 + k0 + g8];
    af.q[1] = *(const uint4*)&Qs[row * 72 + k0 + 16 + g8];
#pragma unroll
    for (int j = 0; j < 4; j++) {
      F16x16 bf;
      int brow = j * 16 + l15;
      bf.q[0] = *(const uint4*)&Ks[brow * 72 + k0 + g8];
      bf.q[1] = *(const uint4*)&Ks[brow * 72 + k0 + 16 + g8];
      s[j] = __builtin_amdgcn_wmma_f32_16x16x32_f16(
          false, af.v, false, bf.v, (short)0, s[j], false, false);
    }
  }

  // ---- softmax over 64 keys ----
  const float scale = 0.125f;  // 1/sqrt(64)
#pragma unroll
  for (int r = 0; r < 8; r++) {
    float mx = fmaxf(fmaxf(s[0][r], s[1][r]), fmaxf(s[2][r], s[3][r]));
    mx = red_max16(mx);
    float sum = 0.f;
#pragma unroll
    for (int j = 0; j < 4; j++) {
      float e = __expf(scale * (s[j][r] - mx));
      s[j][r] = e; sum += e;
    }
    sum = red_sum16(sum);
    float inv = 1.f / sum;
#pragma unroll
    for (int j = 0; j < 4; j++) s[j][r] *= inv;
  }
#pragma unroll
  for (int j = 0; j < 4; j++)
#pragma unroll
    for (int r = 0; r < 8; r++)
      Ps[(rb + g8 + r) * 72 + j * 16 + l15] = (half_t)s[j][r];
  __syncthreads();

  // ---- O = P V (V^T staged) ----
  v8f o[4]; o[0] = zero; o[1] = zero; o[2] = zero; o[3] = zero;
#pragma unroll
  for (int k0 = 0; k0 < 64; k0 += 32) {
    F16x16 af;
    int row = rb + l15;
    af.q[0] = *(const uint4*)&Ps[row * 72 + k0 + g8];
    af.q[1] = *(const uint4*)&Ps[row * 72 + k0 + 16 + g8];
#pragma unroll
    for (int j = 0; j < 4; j++) {
      F16x16 bf;
      int brow = j * 16 + l15;
      bf.q[0] = *(const uint4*)&Vt[brow * 72 + k0 + g8];
      bf.q[1] = *(const uint4*)&Vt[brow * 72 + k0 + 16 + g8];
      o[j] = __builtin_amdgcn_wmma_f32_16x16x32_f16(
          false, af.v, false, bf.v, (short)0, o[j], false, false);
    }
  }
#pragma unroll
  for (int j = 0; j < 4; j++)
#pragma unroll
    for (int r = 0; r < 8; r++) {
      int m = rb + g8 + r;
      int d = j * 16 + l15;
      long qt = q_token(mode, grp, m);
      Og[qt * C_CH + hoff + d] = (half_t)o[j][r];
    }
}

// ---------------------------------------------------------------------------
// host-side orchestration
// ---------------------------------------------------------------------------
extern "C" void kernel_launch(void* const* d_in, const int* in_sizes, int n_in,
                              void* d_out, int out_size, void* d_ws, size_t ws_size,
                              hipStream_t stream) {
  (void)in_sizes; (void)n_in; (void)out_size; (void)ws_size;
  const float* x    = (const float*)d_in[0];
  const float* n1w  = (const float*)d_in[1];
  const float* n1b  = (const float*)d_in[2];
  const float* qkvb = (const float*)d_in[4];
  const float* alb  = (const float*)d_in[6];
  const float* alob = (const float*)d_in[8];
  const float* plb  = (const float*)d_in[10];
  const float* n2w  = (const float*)d_in[11];
  const float* n2b  = (const float*)d_in[12];
  const float* qgb  = (const float*)d_in[14];
  const float* kvb  = (const float*)d_in[16];
  const float* agb  = (const float*)d_in[18];
  const float* agob = (const float*)d_in[20];
  const float* pgb  = (const float*)d_in[22];
  const float* n3w  = (const float*)d_in[23];
  const float* n3b  = (const float*)d_in[24];
  const float* f1b  = (const float*)d_in[26];
  const float* f2b  = (const float*)d_in[28];

  char* ws = (char*)d_ws;
  size_t off = 0;
  auto carve = [&](size_t bytes) -> void* {
    void* p = ws + off;
    off = (off + bytes + 255) & ~(size_t)255;
    return p;
  };
  const long T = T_TOK;
  float*  xr    = (float*)carve((size_t)T * C_CH * 4);
  half_t* x1h   = (half_t*)carve((size_t)T * C_CH * 2);
  half_t* qkvh  = (half_t*)carve((size_t)T * 768 * 2);   // also scoreh/qgh later
  half_t* qkv2h = (half_t*)carve((size_t)T * 768 * 2);   // also kvh/hh later
  half_t* oh    = (half_t*)carve((size_t)T * C_CH * 2);
  half_t* oloc  = (half_t*)carve((size_t)T * C_CH * 2);
  half_t* kgp   = (half_t*)carve((size_t)512 * C_CH * 2);
  half_t* vgp   = (half_t*)carve((size_t)512 * C_CH * 2);
  float*  smap  = (float*)carve((size_t)T * 4);
  int*    gidx  = (int*)carve(512 * 4);
  half_t* qkvw_h = (half_t*)carve(768 * 256 * 2);
  half_t* alw_h  = (half_t*)carve(768 * 256 * 2);
  half_t* alow_h = (half_t*)carve(256 * 256 * 2);
  half_t* plw_h  = (half_t*)carve(256 * 256 * 2);
  half_t* qgw_h  = (half_t*)carve(256 * 256 * 2);
  half_t* kvw_h  = (half_t*)carve(512 * 256 * 2);
  half_t* agw_h  = (half_t*)carve(768 * 256 * 2);
  half_t* agow_h = (half_t*)carve(256 * 256 * 2);
  half_t* pgw_h  = (half_t*)carve(256 * 256 * 2);
  half_t* f1w_h  = (half_t*)carve(512 * 256 * 2);
  half_t* f2w_h  = (half_t*)carve(256 * 512 * 2);

  half_t* scoreh = qkvh;
  half_t* qgh    = qkvh + (size_t)T * C_CH;
  half_t* kvh    = qkv2h;           // [T,512]
  half_t* hh     = qkv2h;           // [T,512], reused after kvh is done

  (void)hipMemcpyAsync(xr, x, (size_t)T * C_CH * 4, hipMemcpyDeviceToDevice, stream);

  auto pack = [&](int idx, half_t* dst, int n) {
    f2h_kernel<<<(n + 255) / 256, 256, 0, stream>>>((const float*)d_in[idx], dst, n);
  };
  pack(3,  qkvw_h, 768 * 256);
  pack(5,  alw_h,  768 * 256);
  pack(7,  alow_h, 256 * 256);
  pack(9,  plw_h,  256 * 256);
  pack(13, qgw_h,  256 * 256);
  pack(15, kvw_h,  512 * 256);
  pack(17, agw_h,  768 * 256);
  pack(19, agow_h, 256 * 256);
  pack(21, pgw_h,  256 * 256);
  pack(25, f1w_h,  512 * 256);
  pack(27, f2w_h,  256 * 512);

  dim3 blk(256);

  // ===== stage A: local windowed attention =====
  ln_kernel<<<T / 64, blk, 0, stream>>>(x, n1w, n1b, x1h);
  gemm_kernel<0><<<dim3(T / BM, 768 / BN), blk, 0, stream>>>(
      x1h, 256, qkvw_h, 256, qkvb, qkvh, 768, nullptr, nullptr, nullptr, 256);
  for (int s3 = 0; s3 < 3; s3++) {
    gemm_kernel<0><<<dim3(T / BM, 256 / BN), blk, 0, stream>>>(
        qkvh + s3 * 256, 768, alw_h + (size_t)s3 * 256 * 256, 256, alb + s3 * 256,
        qkv2h + s3 * 256, 768, nullptr, nullptr, nullptr, 256);
  }
  attn_kernel<<<2048, 128, 0, stream>>>(qkv2h, 768, qkv2h + 256, 768,
                                        qkv2h + 512, 768, oh, 0);
  gemm_kernel<0><<<dim3(T / BM, 256 / BN), blk, 0, stream>>>(
      oh, 256, alow_h, 256, alob, oloc, 256, nullptr, nullptr, nullptr, 256);
  gemm_kernel<2><<<dim3(T / BM, 256 / BN), blk, 0, stream>>>(
      oloc, 256, plw_h, 256, plb, nullptr, 0, xr, nullptr, nullptr, 256);

  // ===== stage B: top-k routed global attention =====
  ln_kernel<<<T / 64, blk, 0, stream>>>(xr, n2w, n2b, x1h);
  gemm_kernel<0><<<dim3(T / BM, 256 / BN), blk, 0, stream>>>(
      x1h, 256, qgw_h, 256, qgb, scoreh, 256, nullptr, nullptr, nullptr, 256);
  sqsum_kernel<<<T / 8, blk, 0, stream>>>(scoreh, smap);
  gemm_kernel<0><<<dim3(T / BM, 512 / BN), blk, 0, stream>>>(
      x1h, 256, kvw_h, 256, kvb, kvh, 512, nullptr, nullptr, nullptr, 256);
  topk_kernel<<<8, blk, 0, stream>>>(smap, gidx);
  gemm_kernel<0><<<dim3(512 / BM, 256 / BN), blk, 0, stream>>>(
      kvh, 512, agw_h + (size_t)256 * 256, 256, agb + 256,
      kgp, 256, nullptr, nullptr, gidx, 256);
  gemm_kernel<0><<<dim3(512 / BM, 256 / BN), blk, 0, stream>>>(
      kvh + 256, 512, agw_h + (size_t)512 * 256, 256, agb + 512,
      vgp, 256, nullptr, nullptr, gidx, 256);
  gemm_kernel<0><<<dim3(T / BM, 256 / BN), blk, 0, stream>>>(
      scoreh, 256, agw_h, 256, agb, qgh, 256, nullptr, nullptr, nullptr, 256);
  attn_kernel<<<2048, 128, 0, stream>>>(qgh, 256, kgp, 256, vgp, 256, oh, 1);
  gemm_kernel<0><<<dim3(T / BM, 256 / BN), blk, 0, stream>>>(
      oh, 256, agow_h, 256, agob, oloc, 256, nullptr, nullptr, nullptr, 256);
  gemm_kernel<2><<<dim3(T / BM, 256 / BN), blk, 0, stream>>>(
      oloc, 256, pgw_h, 256, pgb, nullptr, 0, xr, nullptr, nullptr, 256);

  // ===== stage C: FFN =====
  ln_kernel<<<T / 64, blk, 0, stream>>>(xr, n3w, n3b, x1h);
  gemm_kernel<1><<<dim3(T / BM, 512 / BN), blk, 0, stream>>>(
      x1h, 256, f1w_h, 256, f1b, hh, 512, nullptr, nullptr, nullptr, 256);
  gemm_kernel<3><<<dim3(T / BM, 256 / BN), blk, 0, stream>>>(
      hh, 512, f2w_h, 512, f2b, nullptr, 0, xr, (float*)d_out, nullptr, 512);
}
